// VectorQuantizer_89275190215554
// MI455X (gfx1250) — matrix operand order repaired and checked by reading, not compiled, not run on hardware
//
#include <hip/hip_runtime.h>
#include <hip/hip_bf16.h>
#include <math.h>

// ---- problem constants -----------------------------------------------------
#define BATCH   64
#define DIM     64
#define TLEN    4096
#define KCODES  512
#define NROWS   (BATCH * TLEN)                 // 262144
#define OUT_N   ((size_t)BATCH * DIM * TLEN)   // 16777216
#define LOSS_OFF (OUT_N)
#define IDX_OFF  (OUT_N + 1)
#define PERP_OFF (OUT_N + 1 + (size_t)NROWS)

#define PAD 72   // halfs per LDS row (64 + 8 pad)

typedef _Float16 v16h __attribute__((ext_vector_type(16)));
typedef _Float16 v8h  __attribute__((ext_vector_type(8)));
typedef float    v8f  __attribute__((ext_vector_type(8)));

// ---- kernel 1: embedding -> f16 table, ||e||^2, zero accumulators ----------
__global__ __launch_bounds__(32)
void vq_prep(const float* __restrict__ emb, float* counts, float* sse,
             float* enorm, _Float16* emb16) {
    const int k    = blockIdx.x;    // code id, 0..511
    const int lane = threadIdx.x;   // 0..31
    float e0 = emb[k * DIM + lane];
    float e1 = emb[k * DIM + 32 + lane];
    emb16[k * DIM + lane]      = (_Float16)e0;
    emb16[k * DIM + 32 + lane] = (_Float16)e1;
    float s = e0 * e0 + e1 * e1;
#pragma unroll
    for (int m = 16; m >= 1; m >>= 1) s += __shfl_xor(s, m, 32);
    if (lane == 0) { enorm[k] = s; counts[k] = 0.0f; }
    if (k == 0 && lane == 1) { *sse = 0.0f; }
}

// ---- kernel 2: d2 = ||e||^2 - 2 x.e entirely inside WMMA + argmin ----------
__global__ __launch_bounds__(256)
void vq_main(const float* __restrict__ x,        // [B, D, T] f32
             const float* __restrict__ embf32,   // [K, D] f32
             const _Float16* __restrict__ emb16, // [K, D] f16
             const float* __restrict__ enorm,    // [K] f32
             float* counts, float* sse,
             float* __restrict__ out)            // d_out base (float)
{
    __shared__ _Float16 xs[128 * PAD];   // (-2*x) tile, f16, [t_local][d]
    __shared__ float    en_s[KCODES];
    __shared__ int      idx_s[128];

    const int bbb = blockIdx.x >> 5;          // batch index
    const int t0  = (blockIdx.x & 31) << 7;   // first t of tile (128 rows)
    const int tid = threadIdx.x;

    // ||e||^2 -> LDS
    en_s[tid]       = enorm[tid];
    en_s[tid + 256] = enorm[tid + 256];

    // stage x tile: coalesced f32 reads along t; store f16(-2x) transposed
#pragma unroll
    for (int it = 0; it < 8; ++it) {
        const int lin = it * 256 + tid;
        const int d   = lin >> 5;
        const int t4  = (lin & 31) << 2;
        const float4 v = *(const float4*)(x + ((size_t)bbb * DIM + d) * TLEN + t0 + t4);
        xs[(t4 + 0) * PAD + d] = (_Float16)(-2.0f * v.x);
        xs[(t4 + 1) * PAD + d] = (_Float16)(-2.0f * v.y);
        xs[(t4 + 2) * PAD + d] = (_Float16)(-2.0f * v.z);
        xs[(t4 + 3) * PAD + d] = (_Float16)(-2.0f * v.w);
    }
    __syncthreads();

    const int wid  = tid >> 5;
    const int lane = tid & 31;
    const int m    = lane & 15;   // A row / B column within tile
    const int half = lane >> 4;
    const int mrow = wid << 4;    // this wave's 16-row strip

    // A fragments (16x32 f16 layout: lane l holds row l&15, two 8-half runs)
    v16h a0, a1;
    {
        const _Float16* rp = &xs[(mrow + m) * PAD + 16 * half];
        v8h lo0 = *(const v8h*)(rp);        // d = 16h .. 16h+7     (K chunk 0)
        v8h hi0 = *(const v8h*)(rp + 8);    // d = 16h+8 .. 16h+15
        v8h lo1 = *(const v8h*)(rp + 32);   // same, K chunk 1 (d0 = 32)
        v8h hi1 = *(const v8h*)(rp + 40);
#pragma unroll
        for (int j = 0; j < 8; ++j) {
            a0[j] = lo0[j]; a0[j + 8] = hi0[j];
            a1[j] = lo1[j]; a1[j + 8] = hi1[j];
        }
    }

    float best[8];
    int   bidx[8];
#pragma unroll
    for (int i = 0; i < 8; ++i) { best[i] = 3.4e38f; bidx[i] = 0; }

    // software-pipelined sweep of 32 code tiles of 16 codes each
    int c = m;                                 // kt=0 code for this lane
    v16h b0 = *(const v16h*)(emb16 + c * DIM + 16 * half);
    v16h b1 = *(const v16h*)(emb16 + c * DIM + 32 + 16 * half);
    float en = en_s[c];

    for (int kt = 0; kt < 32; ++kt) {
        const v16h cb0 = b0, cb1 = b1;
        const int  cc  = c;
        v8f acc = { en, en, en, en, en, en, en, en };   // start from ||e||^2

        // prefetch next tile (wrapped index on last iter; result unused)
        c = (c + 16) & (KCODES - 1);
        b0 = *(const v16h*)(emb16 + c * DIM + 16 * half);
        b1 = *(const v16h*)(emb16 + c * DIM + 32 + 16 * half);
        en = en_s[c];

        acc = __builtin_amdgcn_wmma_f32_16x16x32_f16(false, a0, false, cb0,
                                                     (short)0, acc, false, false);
        acc = __builtin_amdgcn_wmma_f32_16x16x32_f16(false, a1, false, cb1,
                                                     (short)0, acc, false, false);
        // acc[i] == ||e||^2 - 2 x.e  (distance, up to the common ||x||^2 term)
#pragma unroll
        for (int i = 0; i < 8; ++i) {
            if (acc[i] < best[i]) { best[i] = acc[i]; bidx[i] = cc; }
        }
    }

    // reduce argmin across the 16 lanes sharing each C row (first-idx ties)
#pragma unroll
    for (int s = 1; s < 16; s <<= 1) {
#pragma unroll
        for (int i = 0; i < 8; ++i) {
            const float ov = __shfl_xor(best[i], s, 16);
            const int   oi = __shfl_xor(bidx[i], s, 16);
            if (ov < best[i] || (ov == best[i] && oi < bidx[i])) {
                best[i] = ov; bidx[i] = oi;
            }
        }
    }
    if (m == 0) {
#pragma unroll
        for (int i = 0; i < 8; ++i) {
            const int row = mrow + 8 * half + i;   // C row mapping
            idx_s[row] = bidx[i];
            atomicAdd(&counts[bidx[i]], 1.0f);
            out[IDX_OFF + (size_t)bbb * TLEN + t0 + row] = (float)bidx[i];
        }
    }
    __syncthreads();

    // phase 3: quantized output (coalesced [B,D,T] stores) + loss SSE (f32 x)
    float lacc = 0.0f;
#pragma unroll
    for (int it = 0; it < 8; ++it) {
        const int lin = it * 256 + tid;
        const int d   = lin >> 5;
        const int t4  = (lin & 31) << 2;
        const size_t gb = ((size_t)bbb * DIM + d) * TLEN + t0 + t4;
        const float4 v = *(const float4*)(x + gb);
        float4 q;
        q.x = embf32[idx_s[t4 + 0] * DIM + d];
        q.y = embf32[idx_s[t4 + 1] * DIM + d];
        q.z = embf32[idx_s[t4 + 2] * DIM + d];
        q.w = embf32[idx_s[t4 + 3] * DIM + d];
        *(float4*)(out + gb) = q;
        const float dx = q.x - v.x, dy = q.y - v.y;
        const float dz = q.z - v.z, dw = q.w - v.w;
        lacc += dx * dx + dy * dy + dz * dz + dw * dw;
    }
#pragma unroll
    for (int s = 16; s >= 1; s >>= 1) lacc += __shfl_xor(lacc, s, 32);
    if (lane == 0) atomicAdd(sse, lacc);
}

// ---- kernel 3: loss + perplexity scalars -----------------------------------
__global__ __launch_bounds__(512)
void vq_fin(const float* __restrict__ counts, const float* __restrict__ sse,
            float* __restrict__ out) {
    __shared__ float red[16];
    const int tid = threadIdx.x;             // 0..511 == code id
    const float p = counts[tid] * (1.0f / (float)NROWS);
    float s = p * logf(p + 1e-10f);
#pragma unroll
    for (int m = 16; m >= 1; m >>= 1) s += __shfl_xor(s, m, 32);
    if ((tid & 31) == 0) red[tid >> 5] = s;
    __syncthreads();
    if (tid < 16) {
        float v = red[tid];
#pragma unroll
        for (int m = 8; m >= 1; m >>= 1) v += __shfl_xor(v, m, 16);
        if (tid == 0) {
            out[PERP_OFF] = expf(-v);
            out[LOSS_OFF] = 1.25f * (*sse) * (1.0f / (float)OUT_N);
        }
    }
}

// ---- launcher ---------------------------------------------------------------
extern "C" void kernel_launch(void* const* d_in, const int* in_sizes, int n_in,
                              void* d_out, int out_size, void* d_ws, size_t ws_size,
                              hipStream_t stream) {
    const float* x   = (const float*)d_in[0];   // [64, 64, 4096] f32
    const float* emb = (const float*)d_in[1];   // [512, 64] f32
    float* out = (float*)d_out;

    // workspace carve (floats): counts[512] | sse | pad | enorm[512] @768 | emb16 @2048
    float*    ws     = (float*)d_ws;
    float*    counts = ws;
    float*    sse    = ws + 512;
    float*    enorm  = ws + 768;
    _Float16* emb16  = (_Float16*)(ws + 2048);  // 64 KB

    vq_prep<<<KCODES, 32, 0, stream>>>(emb, counts, sse, enorm, emb16);
    vq_main<<<(NROWS / 128), 256, 0, stream>>>(x, emb, emb16, enorm, counts, sse, out);
    vq_fin<<<1, 512, 0, stream>>>(counts, sse, out);
}
